// MambaBlock_28578712387731
// MI455X (gfx1250) — matrix operand order repaired and checked by reading, not compiled, not run on hardware
//
#include <hip/hip_runtime.h>
#include <hip/hip_bf16.h>

// ---------------------------------------------------------------------------
// Mamba block forward for MI455X (gfx1250, wave32, WMMA + TDM).
//  - All GEMM operands bf16 in memory (one-time weight converts; activation
//    producers emit bf16 mirrors), so K-panel staging is a pure 2-D tile copy
//    done by the Tensor Data Mover with hardware edge zero-fill and hardware
//    LDS row padding (pad_interval=16 dwords, amount=1 dword -> 34-elem rows).
//  - TDM is double-buffered: panel k+1 is issued before waiting on panel k
//    (TENSORcnt in-order per wave => s_wait_tensorcnt 2 == panel k arrived).
//  - GEMM core: v_wmma_f32_16x16x32_bf16, 128x128x32 block tiles, 8 waves;
//    all fragment ds_loads issued before the WMMA chain for ILP.
//  - Scan: one channel per 16-lane half-wave, __shfl_xor N=16 reduction.
// ---------------------------------------------------------------------------

#define L_   2048
#define DM_  768
#define DE_  1536
#define DR_  48
#define NS_  16
#define KC_  4
#define N_IN  (2 * DE_)         // 3072
#define N_DTBC (DR_ + 2 * NS_)  // 80

typedef __attribute__((ext_vector_type(16))) __bf16 v16bf;
typedef __attribute__((ext_vector_type(8)))  float  v8f;
typedef __attribute__((ext_vector_type(4)))  unsigned int v4u;
typedef __attribute__((ext_vector_type(8)))  int v8i_t;
typedef __attribute__((ext_vector_type(4)))  int v4i_t;

#if __has_builtin(__builtin_amdgcn_tensor_load_to_lds) && \
    __has_builtin(__builtin_amdgcn_s_wait_tensorcnt)
#define HAVE_TDM 1
#else
#define HAVE_TDM 0
#endif

__device__ __forceinline__ __bf16 f2bf(float f) {
  unsigned u = __builtin_bit_cast(unsigned, f);
  u += 0x7FFFu + ((u >> 16) & 1u);              // round-to-nearest-even
  return __builtin_bit_cast(__bf16, (unsigned short)(u >> 16));
}

__device__ __forceinline__ float silu_f(float x) {
  return x / (1.f + __expf(-x));
}

// ---------------------------------------------------------------------------
// GEMM tiling
// ---------------------------------------------------------------------------
#define BM 128
#define BN 128
#define BK 32
#define SROW (BK + 2)   // 34 bf16 = 17 dwords; produced in HW by TDM padding

#if HAVE_TDM
// D# (ISA cdna5 §8) for a 2-D bf16 tile load: tile = 128 rows x 32 elems,
// row stride = stride_elems, zero-fill beyond (rem_k, rem_rows), LDS pad:
// 1 dword inserted every 16 dwords (=> 34-elem LDS row stride).
__device__ __forceinline__ void tdm_load_2d(unsigned lds_off,
                                            const __bf16* gptr,
                                            unsigned rem_k, unsigned rem_rows,
                                            unsigned stride_elems) {
  unsigned long long ga = (unsigned long long)(uintptr_t)gptr;
  v4u g0;
  g0[0] = 1u;                                           // count=1, user D#
  g0[1] = lds_off;                                      // lds_addr (bytes)
  g0[2] = (unsigned)(ga & 0xFFFFFFFFu);                 // global_addr lo
  g0[3] = (unsigned)((ga >> 32) & 0x01FFFFFFu) | (2u << 30);  // addr hi | type=2
  v8i_t g1;
  g1[0] = (int)((1u << 16)      // data_size = 1 -> 2 bytes
              | (1u << 20)      // pad_enable
              | (3u << 22)      // pad_interval: 16 dwords
              | (0u << 25));    // pad_amount: 1 dword
  g1[1] = (int)((rem_k & 0xFFFFu) << 16);                          // dim0 lo16
  g1[2] = (int)(((rem_k >> 16) & 0xFFFFu) | ((rem_rows & 0xFFFFu) << 16));
  g1[3] = (int)(((rem_rows >> 16) & 0xFFFFu) | ((unsigned)BK << 16)); // tile_dim0
  g1[4] = (int)(unsigned)BM;                                       // tile_dim1
  g1[5] = (int)stride_elems;                                       // dim0_stride lo
  g1[6] = 0;
  g1[7] = 0;
  v4i_t z4 = {0, 0, 0, 0};
  v8i_t z8 = {0, 0, 0, 0, 0, 0, 0, 0};
  __builtin_amdgcn_tensor_load_to_lds(g0, g1, z4, z4, z8, 0);
}
#endif

// out[M,N] = A[M,K](bf16, stride lda) * W[N,K](bf16, stride ldw)^T + bias
// ACT: 0 = none, 1 = softplus.  Optional bf16 mirror output (outb).
template <int ACT>
__global__ __launch_bounds__(256) void gemm_bf16_wmma(
    const __bf16* __restrict__ A, int lda,
    const __bf16* __restrict__ W, int ldw,
    const float* __restrict__ bias,
    float* __restrict__ out, __bf16* __restrict__ outb, int ldo,
    int M, int N, int K)
{
  __shared__ __bf16 sA[2][BM * SROW];
  __shared__ __bf16 sB[2][BN * SROW];

  const int t      = threadIdx.x;
  const int lane   = t & 31;
  const int wave   = t >> 5;
  const int waveM  = wave & 3;
  const int waveN  = wave >> 2;
  const int blockM = blockIdx.x * BM;
  const int blockN = blockIdx.y * BN;

  v8f acc[2][4];
  #pragma unroll
  for (int mt = 0; mt < 2; ++mt)
    #pragma unroll
    for (int nt = 0; nt < 4; ++nt)
      #pragma unroll
      for (int v = 0; v < 8; ++v)
        acc[mt][nt][v] = 0.f;

  const __bf16* Ab = A + (long)blockM * lda;
  const __bf16* Wb = W + (long)blockN * ldw;
  const unsigned remM = (unsigned)(M - blockM);
  const unsigned remN = (unsigned)(N - blockN);
  const int nk = (K + BK - 1) / BK;

#if HAVE_TDM
  if (wave == 0) {   // prologue: panel 0 -> buffer 0
    tdm_load_2d((unsigned)(uintptr_t)&sA[0][0], Ab, (unsigned)K, remM, (unsigned)lda);
    tdm_load_2d((unsigned)(uintptr_t)&sB[0][0], Wb, (unsigned)K, remN, (unsigned)ldw);
  }
#endif

  for (int ik = 0; ik < nk; ++ik) {
    const int k0  = ik * BK;
    const int cur = ik & 1;
#if HAVE_TDM
    if (wave == 0) {
      if (ik + 1 < nk) {
        // issue panel ik+1 into the other buffer, then wait with slack 2:
        // TENSORcnt completes in order, so <=2 outstanding means panel ik
        // (the 2 older loads) has fully landed in LDS.
        const int k1 = k0 + BK;
        tdm_load_2d((unsigned)(uintptr_t)&sA[cur ^ 1][0], Ab + k1,
                    (unsigned)(K - k1), remM, (unsigned)lda);
        tdm_load_2d((unsigned)(uintptr_t)&sB[cur ^ 1][0], Wb + k1,
                    (unsigned)(K - k1), remN, (unsigned)ldw);
        __builtin_amdgcn_s_wait_tensorcnt(2);
      } else {
        __builtin_amdgcn_s_wait_tensorcnt(0);
      }
    }
    __syncthreads();   // publish panel ik to all waves
#else
    __syncthreads();   // previous readers done before overwrite
    {
      const int prow = t >> 1;
      const int pcol = (t & 1) * 16;
      const __bf16* ap = A + (long)(blockM + prow) * lda;
      const int gn = blockN + prow;
      const __bf16* wp = W + (long)gn * ldw;
      #pragma unroll
      for (int i = 0; i < 16; ++i) {
        const int kk = k0 + pcol + i;
        __bf16 zb = __builtin_bit_cast(__bf16, (unsigned short)0);
        sA[cur][prow * SROW + pcol + i] = (kk < K) ? ap[kk] : zb;
        sB[cur][prow * SROW + pcol + i] = (gn < N && kk < K) ? wp[kk] : zb;
      }
    }
    __syncthreads();
#endif

    // WMMA fragments, wave32 layouts (CDNA5 ISA §7.12.2).
    // Load ALL fragments (2 A + 4 B) into distinct registers first so the
    // ds_loads pipeline and the 8 WMMAs issue back-to-back.
    const int mlane  = lane & 15;
    const int khalfA = (lane >> 4) * 8;
    const int khB    = (lane >> 4) * 16;
    const __bf16* sAc = &sA[cur][0];
    const __bf16* sBc = &sB[cur][0];

    v16bf afrag[2];
    #pragma unroll
    for (int mt = 0; mt < 2; ++mt) {
      const __bf16* ap = &sAc[(waveM * 32 + mt * 16 + mlane) * SROW];
      #pragma unroll
      for (int i = 0; i < 4; ++i) {
        afrag[mt][2 * i]         = ap[khalfA + 2 * i];
        afrag[mt][2 * i + 1]     = ap[khalfA + 2 * i + 1];
        afrag[mt][8 + 2 * i]     = ap[16 + khalfA + 2 * i];
        afrag[mt][8 + 2 * i + 1] = ap[16 + khalfA + 2 * i + 1];
      }
    }
    v16bf bfrag[4];
    #pragma unroll
    for (int nt = 0; nt < 4; ++nt) {
      const __bf16* bp = &sBc[(waveN * 64 + nt * 16 + mlane) * SROW];
      #pragma unroll
      for (int i = 0; i < 8; ++i) {
        bfrag[nt][2 * i]     = bp[khalfA + 2 * i + (i >= 4 ? 8 : 0)];
        bfrag[nt][2 * i + 1] = bp[khalfA + 2 * i + 1 + (i >= 4 ? 8 : 0)];
      }
    }
    #pragma unroll
    for (int nt = 0; nt < 4; ++nt) {
      acc[0][nt] = __builtin_amdgcn_wmma_f32_16x16x32_bf16(
          false, afrag[0], false, bfrag[nt], (short)0, acc[0][nt], false, false);
      acc[1][nt] = __builtin_amdgcn_wmma_f32_16x16x32_bf16(
          false, afrag[1], false, bfrag[nt], (short)0, acc[1][nt], false, false);
    }
    __syncthreads();   // readers done; buffer may be re-filled next iteration
  }

  // epilogue: C/D row = vgpr + 8*(lane>>4), col = lane&15
  const int colBase = blockN + waveN * 64 + (lane & 15);
  const int rowHalf = (lane >> 4) * 8;
  #pragma unroll
  for (int mt = 0; mt < 2; ++mt) {
    #pragma unroll
    for (int nt = 0; nt < 4; ++nt) {
      const int col = colBase + nt * 16;
      if (col < N) {
        const float bv = bias[col];
        const int row0 = blockM + waveM * 32 + mt * 16 + rowHalf;
        #pragma unroll
        for (int v = 0; v < 8; ++v) {
          float r = acc[mt][nt][v] + bv;
          if (ACT == 1) r = (r > 20.f) ? r : log1pf(__expf(r));
          out[(long)(row0 + v) * ldo + col] = r;
          if (outb) outb[(long)(row0 + v) * ldo + col] = f2bf(r);
        }
      }
    }
  }
}

// ---------------------------------------------------------------------------
// one-time f32 -> bf16 conversion (weights / input)
// ---------------------------------------------------------------------------
__global__ __launch_bounds__(256) void f2bf_kernel(
    const float* __restrict__ src, __bf16* __restrict__ dst, int n) {
  const int i = blockIdx.x * blockDim.x + threadIdx.x;
  if (i < n) dst[i] = f2bf(src[i]);
}

// ---------------------------------------------------------------------------
// causal depthwise conv (K=4) + bias + SiLU; emits f32 (scan) + bf16 (GEMM A)
// ---------------------------------------------------------------------------
__global__ __launch_bounds__(256) void conv_silu_kernel(
    const float* __restrict__ xz,      // (L, 2*DE), xh = cols [0, DE)
    const float* __restrict__ conv_w,  // (DE, 1, 4)
    const float* __restrict__ conv_b,  // (DE)
    float* __restrict__ xc,            // (L, DE) f32
    __bf16* __restrict__ xcb)          // (L, DE) bf16 mirror
{
  const int idx = blockIdx.x * blockDim.x + threadIdx.x;
  if (idx >= L_ * DE_) return;
  const int d = idx % DE_;
  const int l = idx / DE_;
  float s = conv_b[d];
  #pragma unroll
  for (int k = 0; k < KC_; ++k) {
    const int tt = l - (KC_ - 1) + k;
    if (tt >= 0) s += conv_w[d * KC_ + k] * xz[(long)tt * N_IN + d];
  }
  const float r = silu_f(s);
  xc[idx]  = r;
  xcb[idx] = f2bf(r);
}

// ---------------------------------------------------------------------------
// selective scan + gate; 16-lane half-wave per channel, lane n = state index.
// Emits bf16 directly (only out_proj consumes it).
// ---------------------------------------------------------------------------
__global__ __launch_bounds__(256) void scan_gate_kernel(
    const float* __restrict__ dtBC,   // (L, 80): [48:64)=B, [64:80)=C
    const float* __restrict__ dtf,    // (L, DE)
    const float* __restrict__ A_log,  // (DE, 16)
    const float* __restrict__ xc,     // (L, DE)
    const float* __restrict__ xz,     // (L, 2*DE): z = cols [DE, 2*DE)
    __bf16* __restrict__ ygb)         // (L, DE) bf16
{
  const int lane = threadIdx.x & 31;
  const int n    = lane & 15;
  const int half = lane >> 4;
  const int wave = threadIdx.x >> 5;
  const int d    = blockIdx.x * 16 + wave * 2 + half;

  const float a = -__expf(A_log[d * NS_ + n]);
  float h = 0.f;

  for (int l = 0; l < L_; ++l) {
    const float dt = dtf[(long)l * DE_ + d];
    const float xv = xc [(long)l * DE_ + d];
    const float bn = dtBC[(long)l * N_DTBC + DR_ + n];
    const float cn = dtBC[(long)l * N_DTBC + DR_ + NS_ + n];
    h = __expf(dt * a) * h + dt * bn * xv;
    float y = h * cn;
    y += __shfl_xor(y, 1);
    y += __shfl_xor(y, 2);
    y += __shfl_xor(y, 4);
    y += __shfl_xor(y, 8);
    if (n == 0) {
      const float z = xz[(long)l * N_IN + DE_ + d];
      ygb[(long)l * DE_ + d] = f2bf(y * silu_f(z));
    }
  }
}

// ---------------------------------------------------------------------------
extern "C" void kernel_launch(void* const* d_in, const int* in_sizes, int n_in,
                              void* d_out, int out_size, void* d_ws, size_t ws_size,
                              hipStream_t stream) {
  const float* x      = (const float*)d_in[0];
  const float* W_in   = (const float*)d_in[1];
  const float* b_in   = (const float*)d_in[2];
  const float* conv_w = (const float*)d_in[3];
  const float* conv_b = (const float*)d_in[4];
  const float* Wx     = (const float*)d_in[5];
  const float* bx     = (const float*)d_in[6];
  const float* Wdt    = (const float*)d_in[7];
  const float* bdt    = (const float*)d_in[8];
  const float* A_log  = (const float*)d_in[9];
  const float* W_out  = (const float*)d_in[10];
  const float* b_out  = (const float*)d_in[11];
  float* out = (float*)d_out;

  // workspace partition
  float* xz    = (float*)d_ws;                       // L x 3072
  float* xc    = xz   + (size_t)L_ * N_IN;           // L x 1536
  float* dtBC  = xc   + (size_t)L_ * DE_;            // L x 80
  float* dtf   = dtBC + (size_t)L_ * N_DTBC;         // L x 1536
  __bf16* xb      = (__bf16*)(dtf + (size_t)L_ * DE_);
  __bf16* Wb_in   = xb      + (size_t)L_ * DM_;
  __bf16* Wb_x    = Wb_in   + (size_t)N_IN * DM_;
  __bf16* Wb_dt   = Wb_x    + (size_t)N_DTBC * DE_;
  __bf16* Wb_out  = Wb_dt   + (size_t)DE_ * DR_;
  __bf16* xcb     = Wb_out  + (size_t)DM_ * DE_;
  __bf16* dtBCb   = xcb     + (size_t)L_ * DE_;
  __bf16* ygb     = dtBCb   + (size_t)L_ * N_DTBC;

  // 0) one-time f32 -> bf16 converts
  auto cvt = [&](const float* s, __bf16* d, int n) {
    f2bf_kernel<<<(n + 255) / 256, 256, 0, stream>>>(s, d, n);
  };
  cvt(x,     xb,     L_ * DM_);
  cvt(W_in,  Wb_in,  N_IN * DM_);
  cvt(Wx,    Wb_x,   N_DTBC * DE_);
  cvt(Wdt,   Wb_dt,  DE_ * DR_);
  cvt(W_out, Wb_out, DM_ * DE_);

  // 1) in_proj: xz = x @ W_in^T + b_in  (2048 x 3072, K=768)
  gemm_bf16_wmma<0><<<dim3(L_ / BM, N_IN / BN), 256, 0, stream>>>(
      xb, DM_, Wb_in, DM_, b_in, xz, (__bf16*)nullptr, N_IN, L_, N_IN, DM_);

  // 2) causal depthwise conv + SiLU
  conv_silu_kernel<<<(L_ * DE_ + 255) / 256, 256, 0, stream>>>(
      xz, conv_w, conv_b, xc, xcb);

  // 3) x_proj: dtBC = xc @ Wx^T + bx  (2048 x 80, K=1536), + bf16 mirror
  gemm_bf16_wmma<0><<<dim3(L_ / BM, (N_DTBC + BN - 1) / BN), 256, 0, stream>>>(
      xcb, DE_, Wb_x, DE_, bx, dtBC, dtBCb, N_DTBC, L_, N_DTBC, DE_);

  // 4) dt_proj: dtf = softplus(dtBC[:, :48] @ Wdt^T + bdt)  (2048 x 1536, K=48)
  gemm_bf16_wmma<1><<<dim3(L_ / BM, DE_ / BN), 256, 0, stream>>>(
      dtBCb, N_DTBC, Wb_dt, DR_, bdt, dtf, (__bf16*)nullptr, DE_, L_, DE_, DR_);

  // 5) selective scan + gate -> bf16
  scan_gate_kernel<<<DE_ / 16, 256, 0, stream>>>(
      dtBC, dtf, A_log, xc, xz, ygb);

  // 6) out_proj: out = yg @ W_out^T + b_out  (2048 x 768, K=1536)
  gemm_bf16_wmma<0><<<dim3(L_ / BM, DM_ / BN), 256, 0, stream>>>(
      ygb, DE_, Wb_out, DE_, b_out, out, (__bf16*)nullptr, DM_, L_, DM_, DE_);
}
